// GRU_64441689309678
// MI455X (gfx1250) — matrix operand order repaired and checked by reading, not compiled, not run on hardware
//
#include <hip/hip_runtime.h>

// ---------------- problem constants ----------------
#define SEQ   70
#define BATCH 128
#define EDIM  512
#define HDIM  1024
#define VDIM  10000
#define SB    (SEQ * BATCH)          // 8960 rows in the batched GEMMs
#define KC    64                     // K-chunk staged in LDS per iteration

// ---------------- vector types for WMMA ----------------
typedef __attribute__((ext_vector_type(16))) __bf16 bv16;
typedef __attribute__((ext_vector_type(8)))  __bf16 bv8;
typedef __attribute__((ext_vector_type(8)))  float  fv8;

// pointer types for the async global->LDS builtin: v4i with explicit address spaces
typedef int v4i __attribute__((vector_size(16)));
typedef __attribute__((address_space(1))) v4i* gv4i_p;
typedef __attribute__((address_space(3))) v4i* lv4i_p;

// ---------------- CDNA5 async global->LDS path (probe via __has_builtin) ----------------
#if defined(__has_builtin)
#  if __has_builtin(__builtin_amdgcn_global_load_async_to_lds_b128)
#    define HAVE_ASYNC_LDS 1
#  endif
#  if __has_builtin(__builtin_amdgcn_s_wait_asynccnt)
#    define HAVE_WAIT_ASYNC_BUILTIN 1
#  endif
#endif
#ifndef HAVE_ASYNC_LDS
#  define HAVE_ASYNC_LDS 0
#endif
#ifndef HAVE_WAIT_ASYNC_BUILTIN
#  define HAVE_WAIT_ASYNC_BUILTIN 0
#endif

static __device__ __forceinline__ void wait_async_lds() {
#if HAVE_ASYNC_LDS
#  if HAVE_WAIT_ASYNC_BUILTIN
    __builtin_amdgcn_s_wait_asynccnt(0);
#  else
    asm volatile("s_wait_asynccnt 0" ::: "memory");
#  endif
#endif
}

static __device__ __forceinline__ fv8 wmma_bf16(bv16 a, bv16 b, fv8 c) {
    // D(16x16,f32) = A(16x32,bf16) * B(32x16,bf16) + C
    return __builtin_amdgcn_wmma_f32_16x16x32_bf16(false, a, false, b, (short)0, c, false, false);
}

// A tile (16x32 bf16, row-major source, row stride ld elements).
// Layout: lanes 0-15 hold M=lane, K=0..15 contiguously;
// lanes 16-31 hold M=lane-16, K=16..31 (matches B tile K order).
static __device__ __forceinline__ bv16 load_a_tile(const __bf16* __restrict__ tile, int ld, int lane) {
    const int r  = lane & 15;
    const int kh = (lane >> 4) << 4;          // 0 or 16
    const __bf16* p = tile + (size_t)r * ld + kh;
    bv8 lo = *(const bv8*)(p);                // K = kh .. kh+7
    bv8 hi = *(const bv8*)(p + 8);            // K = kh+8 .. kh+15
    bv16 a;
#pragma unroll
    for (int i = 0; i < 8; ++i) { a[i] = lo[i]; a[8 + i] = hi[i]; }
    return a;
}

// B tile (32x16 bf16) from row-major W (N x K): B[k][n] = W[n][k].
// Layout used here (contiguous K, same as A tile): lane n (0-15) holds K=0..15 contiguously, lane n+16 holds K=16..31.
static __device__ __forceinline__ bv16 load_b_tile(const __bf16* __restrict__ tile, int ld, int lane) {
    const int c  = lane & 15;
    const int ko = (lane >> 4) << 4;          // 0 or 16
    return *(const bv16*)(tile + (size_t)c * ld + ko);
}

static __device__ __forceinline__ float sigmoidf_(float x) {
    return 1.0f / (1.0f + __expf(-x));
}

// ---------------- small utility kernels ----------------
__global__ void cvt_f32_to_bf16(const float* __restrict__ src, __bf16* __restrict__ dst, size_t n) {
    size_t i = (size_t)blockIdx.x * blockDim.x + threadIdx.x;
    size_t stride = (size_t)gridDim.x * blockDim.x;
    for (; i < n; i += stride) dst[i] = (__bf16)src[i];
}

__global__ void embed_gather_bf16(const int* __restrict__ tokens, const float* __restrict__ embW,
                                  __bf16* __restrict__ out) {
    const int sb = blockIdx.x;                      // (s*B + b)
    const int tok = tokens[sb];
    const float* src = embW + (size_t)tok * EDIM;
    __bf16* dst = out + (size_t)sb * EDIM;
    for (int i = threadIdx.x; i < EDIM; i += blockDim.x) dst[i] = (__bf16)src[i];
}

__global__ void init_h_kernel(const float* __restrict__ hsrc, float* __restrict__ hf,
                              __bf16* __restrict__ hbf, int n) {
    int i = blockIdx.x * blockDim.x + threadIdx.x;
    if (i < n) { float v = hsrc[i]; hf[i] = v; hbf[i] = (__bf16)v; }
}

__global__ void copy_f32_kernel(const float* __restrict__ src, float* __restrict__ dst, int n) {
    int i = blockIdx.x * blockDim.x + threadIdx.x;
    if (i < n) dst[i] = src[i];
}

// ---------------- batched GEMM: out(MxN) = X(MxK,bf16) * W(NxK,bf16)^T + bias ----------------
// grid: (Ntiles, M/128), block 256 (8 waves). Each wave: 16 rows x NT*16 cols.
// B panel (NT*16 cols x KC) is staged in LDS (async global->LDS, double buffered);
// all 8 waves of the block share it via ds_load.
template <int NT, typename OutT>
__global__ __launch_bounds__(256)
void gemm_tiled(const __bf16* __restrict__ X, const __bf16* __restrict__ W,
                const float* __restrict__ bias, OutT* __restrict__ out,
                int M, int N, int K, int ncol0) {
    __shared__ __bf16 sB[2][NT * 16 * KC];

    const int tid  = threadIdx.x;
    const int lane = tid & 31;
    const int wave = tid >> 5;
    const int m0   = blockIdx.y * 128 + wave * 16;
    const int n0   = ncol0 + blockIdx.x * (NT * 16);

    const __bf16* A = X + (size_t)m0 * K;

    auto stage = [&](int buf, int kc) {
        const int NCHUNK = NT * 16 * KC / 8;          // 16B chunks in the panel
        for (int c = tid; c < NCHUNK; c += 256) {
            const int row = (c * 8) / KC;             // 0 .. NT*16-1  (output column)
            const int kk  = (c * 8) % KC;
            const __bf16* gp = W + (size_t)(n0 + row) * K + kc + kk;
            __bf16* lp = &sB[buf][row * KC + kk];
#if HAVE_ASYNC_LDS
            __builtin_amdgcn_global_load_async_to_lds_b128((gv4i_p)gp, (lv4i_p)lp, 0, 0);
#else
            *(bv8*)lp = *(const bv8*)gp;
#endif
        }
    };

    fv8 acc[NT];
#pragma unroll
    for (int t = 0; t < NT; ++t) acc[t] = {};

    stage(0, 0);
    int buf = 0;
    for (int kc = 0; kc < K; kc += KC) {
        wait_async_lds();          // my async copies into `buf` have landed in LDS
        __syncthreads();           // everyone's copies landed AND everyone done reading buf^1
        if (kc + KC < K) {
            stage(buf ^ 1, kc + KC);
            __builtin_prefetch(A + kc + KC, 0, 3);
        }
#pragma unroll
        for (int ks = 0; ks < 2; ++ks) {              // KC/32 WMMA K-steps
            bv16 a = load_a_tile(A + kc + ks * 32, K, lane);
#pragma unroll
            for (int t = 0; t < NT; ++t) {
                const __bf16* bp = &sB[buf][(t * 16 + (lane & 15)) * KC +
                                            ks * 32 + ((lane >> 4) << 4)];
                bv16 b = *(const bv16*)bp;            // 2 x ds_load_b128
                acc[t] = wmma_bf16(a, b, acc[t]);
            }
        }
        buf ^= 1;
    }

    const int col = lane & 15;
    const int mb  = m0 + ((lane >> 4) << 3);
#pragma unroll
    for (int t = 0; t < NT; ++t) {
        const int n = n0 + t * 16 + col;
        const float bv = bias ? bias[n] : 0.0f;
#pragma unroll
        for (int v = 0; v < 8; ++v)
            out[(size_t)(mb + v) * N + n] = (OutT)(acc[t][v] + bv);
    }
}

// ---------------- fused GRU recurrent step ----------------
// One launch per timestep. grid: 64 blocks x 256 threads = 512 waves = (B/16)*(H/16) tiles.
// r/z/h-candidate GEMMs share one A tile (3 WMMAs per K-step), then fused gate math.
// Recurrent weights (12 MB total) are L2-resident across the 70 steps.
__global__ __launch_bounds__(256)
void gru_step(const __bf16* __restrict__ hbf, const float* __restrict__ hf,
              const __bf16* __restrict__ Whr, const __bf16* __restrict__ Whz,
              const __bf16* __restrict__ Whh,
              const __bf16* __restrict__ xr, const __bf16* __restrict__ xz,
              const __bf16* __restrict__ xh,
              float* __restrict__ hf_out, __bf16* __restrict__ hbf_out,
              __bf16* __restrict__ ys_out) {
    const int lane = threadIdx.x & 31;
    const int wave = threadIdx.x >> 5;
    const int tile = blockIdx.x * 8 + wave;       // 0..511
    const int m0 = (tile >> 6) * 16;              // batch-row tile (B/16 = 8)
    const int n0 = (tile & 63) * 16;              // hidden-col tile (H/16 = 64)

    fv8 ar = {}, az = {}, ah = {};
    const __bf16* A = hbf + (size_t)m0 * HDIM;

    for (int k = 0; k < HDIM; k += 32) {
        bv16 a = load_a_tile(A + k, HDIM, lane);
        { bv16 b = load_b_tile(Whr + (size_t)n0 * HDIM + k, HDIM, lane); ar = wmma_bf16(a, b, ar); }
        { bv16 b = load_b_tile(Whz + (size_t)n0 * HDIM + k, HDIM, lane); az = wmma_bf16(a, b, az); }
        { bv16 b = load_b_tile(Whh + (size_t)n0 * HDIM + k, HDIM, lane); ah = wmma_bf16(a, b, ah); }
    }

    const int col = n0 + (lane & 15);
    const int mb  = m0 + ((lane >> 4) << 3);
#pragma unroll
    for (int v = 0; v < 8; ++v) {
        const size_t idx = (size_t)(mb + v) * HDIM + col;
        const float r  = sigmoidf_((float)xr[idx] + ar[v]);
        const float z  = sigmoidf_((float)xz[idx] + az[v]);
        const float hc = tanhf((float)xh[idx] + r * ah[v]);
        const float hn = (1.0f - z) * hc + z * hf[idx];
        hf_out[idx]  = hn;
        const __bf16 hb = (__bf16)hn;
        hbf_out[idx] = hb;
        ys_out[idx]  = hb;
    }
}

// ---------------- workspace layout ----------------
static constexpr size_t AL(size_t x) { return (x + 255) & ~(size_t)255; }
static constexpr size_t SZ_EMB  = AL((size_t)SB * EDIM * 2);
static constexpr size_t SZ_X    = AL((size_t)SB * HDIM * 2);     // gate pre-activations, bf16
static constexpr size_t SZ_YS   = AL((size_t)SB * HDIM * 2);
static constexpr size_t SZ_WIN  = AL((size_t)HDIM * EDIM * 2);
static constexpr size_t SZ_WREC = AL((size_t)HDIM * HDIM * 2);
static constexpr size_t SZ_WOUT = AL((size_t)VDIM * HDIM * 2);
static constexpr size_t SZ_HF   = AL((size_t)BATCH * HDIM * 4);
static constexpr size_t SZ_HB   = AL((size_t)BATCH * HDIM * 2);

static constexpr size_t OFF_EMB   = 0;
static constexpr size_t OFF_XR    = OFF_EMB   + SZ_EMB;
static constexpr size_t OFF_XZ    = OFF_XR    + SZ_X;
static constexpr size_t OFF_XH    = OFF_XZ    + SZ_X;
static constexpr size_t OFF_YS0   = OFF_XH    + SZ_X;
static constexpr size_t OFF_YS1   = OFF_YS0   + SZ_YS;
static constexpr size_t OFF_W0IN  = OFF_YS1   + SZ_YS;       // 3 x (H*E)
static constexpr size_t OFF_W0REC = OFF_W0IN  + 3 * SZ_WIN;  // 3 x (H*H)
static constexpr size_t OFF_W1    = OFF_W0REC + 3 * SZ_WREC; // 6 x (H*H)
static constexpr size_t OFF_WOUT  = OFF_W1    + 6 * SZ_WREC;
static constexpr size_t OFF_HF    = OFF_WOUT  + SZ_WOUT;     // 2 x (B*H) f32 ping-pong
static constexpr size_t OFF_HB    = OFF_HF    + 2 * SZ_HF;   // 2 x (B*H) bf16 ping-pong

// ---------------- host-side launch ----------------
extern "C" void kernel_launch(void* const* d_in, const int* in_sizes, int n_in,
                              void* d_out, int out_size, void* d_ws, size_t ws_size,
                              hipStream_t stream) {
    (void)in_sizes; (void)n_in; (void)out_size; (void)ws_size;

    const int*   tokens = (const int*)d_in[0];
    const float* hidden = (const float*)d_in[1];
    const float* embW   = (const float*)d_in[2];
    const float* outW   = (const float*)d_in[3];
    const float* outb   = (const float*)d_in[4];
    // per-layer: [0]=Wir [1]=Wiz [2]=Wih [3]=Whr [4]=Whz [5]=Whh [6]=bir [7]=biz [8]=bih
    const float* lw[2][9];
    for (int l = 0; l < 2; ++l)
        for (int i = 0; i < 9; ++i)
            lw[l][i] = (const float*)d_in[5 + l * 9 + i];

    char* ws = (char*)d_ws;
    __bf16* emb_bf = (__bf16*)(ws + OFF_EMB);
    __bf16* xr     = (__bf16*)(ws + OFF_XR);
    __bf16* xz     = (__bf16*)(ws + OFF_XZ);
    __bf16* xh     = (__bf16*)(ws + OFF_XH);
    __bf16* ys0    = (__bf16*)(ws + OFF_YS0);
    __bf16* ys1    = (__bf16*)(ws + OFF_YS1);
    __bf16 *w0in[3], *w0rec[3], *w1w[6];
    for (int i = 0; i < 3; ++i) w0in[i]  = (__bf16*)(ws + OFF_W0IN  + i * SZ_WIN);
    for (int i = 0; i < 3; ++i) w0rec[i] = (__bf16*)(ws + OFF_W0REC + i * SZ_WREC);
    for (int i = 0; i < 6; ++i) w1w[i]   = (__bf16*)(ws + OFF_W1    + i * SZ_WREC);
    __bf16* woutb = (__bf16*)(ws + OFF_WOUT);
    float*  hf[2]; __bf16* hb[2];
    for (int i = 0; i < 2; ++i) {
        hf[i] = (float*)(ws + OFF_HF + i * SZ_HF);
        hb[i] = (__bf16*)(ws + OFF_HB + i * SZ_HB);
    }
    float* logits = (float*)d_out;
    float* hT_out = logits + (size_t)SB * VDIM;     // (2,B,H) appended

    auto cvt = [&](const float* s, __bf16* d, size_t n) {
        int blocks = (int)((n + 255) / 256); if (blocks > 4096) blocks = 4096;
        cvt_f32_to_bf16<<<blocks, 256, 0, stream>>>(s, d, n);
    };
    // out = X * W^T + bias; main grid uses 64-wide column panels, 16-wide remainder.
    auto gemm = [&](const __bf16* X, const __bf16* W, const float* bias, auto* out,
                    int M, int N, int K) {
        const int nfull = N / 64;
        if (nfull > 0)
            gemm_tiled<4><<<dim3(nfull, M / 128), 256, 0, stream>>>(X, W, bias, out, M, N, K, 0);
        const int rem = N - nfull * 64;               // multiple of 16 for all our shapes
        if (rem > 0)
            gemm_tiled<1><<<dim3(rem / 16, M / 128), 256, 0, stream>>>(X, W, bias, out, M, N, K,
                                                                        nfull * 64);
    };

    // 1) weight conversions to bf16
    for (int i = 0; i < 3; ++i) cvt(lw[0][i],     w0in[i],  (size_t)HDIM * EDIM);
    for (int i = 0; i < 3; ++i) cvt(lw[0][3 + i], w0rec[i], (size_t)HDIM * HDIM);
    for (int i = 0; i < 6; ++i) cvt(lw[1][i],     w1w[i],   (size_t)HDIM * HDIM);
    cvt(outW, woutb, (size_t)VDIM * HDIM);

    // 2) embedding gather -> bf16 activations
    embed_gather_bf16<<<SB, 128, 0, stream>>>(tokens, embW, emb_bf);

    const int BH = BATCH * HDIM;

    for (int layer = 0; layer < 2; ++layer) {
        const __bf16* Xin  = (layer == 0) ? emb_bf : ys0;
        const int     Kin  = (layer == 0) ? EDIM : HDIM;
        const __bf16* Wir  = (layer == 0) ? w0in[0]  : w1w[0];
        const __bf16* Wiz  = (layer == 0) ? w0in[1]  : w1w[1];
        const __bf16* Wih  = (layer == 0) ? w0in[2]  : w1w[2];
        const __bf16* Whr  = (layer == 0) ? w0rec[0] : w1w[3];
        const __bf16* Whz  = (layer == 0) ? w0rec[1] : w1w[4];
        const __bf16* Whh  = (layer == 0) ? w0rec[2] : w1w[5];
        __bf16*       ys   = (layer == 0) ? ys0 : ys1;

        // 3) input-side projections for the whole sequence (big GEMMs + bias), bf16 out
        gemm(Xin, Wir, lw[layer][6], xr, SB, HDIM, Kin);
        gemm(Xin, Wiz, lw[layer][7], xz, SB, HDIM, Kin);
        gemm(Xin, Wih, lw[layer][8], xh, SB, HDIM, Kin);

        // 4) initial hidden state for this layer
        init_h_kernel<<<(BH + 255) / 256, 256, 0, stream>>>(
            hidden + (size_t)layer * BH, hf[0], hb[0], BH);

        // 5) sequential scan, ping-pong buffers (avoids cross-WG WAR hazard)
        for (int t = 0; t < SEQ; ++t) {
            const int cur = t & 1, nxt = cur ^ 1;
            gru_step<<<64, 256, 0, stream>>>(
                hb[cur], hf[cur], Whr, Whz, Whh,
                xr + (size_t)t * BH, xz + (size_t)t * BH, xh + (size_t)t * BH,
                hf[nxt], hb[nxt], ys + (size_t)t * BH);
        }
        // after SEQ=70 steps the final state sits in buffer (SEQ & 1) == 0
        copy_f32_kernel<<<(BH + 255) / 256, 256, 0, stream>>>(
            hf[SEQ & 1], hT_out + (size_t)layer * BH, BH);
    }

    // 6) output projection: logits = ys1 * out_W^T + out_b  (writes d_out directly, f32)
    gemm(ys1, woutb, outb, logits, SB, VDIM, HDIM);
}
